// NewCNNEncoder_12558484373820
// MI455X (gfx1250) — matrix operand order repaired and checked by reading, not compiled, not run on hardware
//
#include <hip/hip_runtime.h>
#include <hip/hip_bf16.h>

// ---------------------------------------------------------------------------
// CDNA5 (gfx1250) implementation of NewCNNEncoder forward.
//   k0_prep : fp32 weights -> padded f16 weight matrices (WMMA-B layout)
//   k1      : one-hot gather (depthwise convs) + 3 pointwise GEMMs via WMMA
//             -> cat[B][960] f16 (l-major: k = l*64 + c), lives in L2 (126MB)
//   k3      : final GEMM [B,960]x[960,1600] via WMMA, staged with CDNA5
//             GLOBAL_LOAD_ASYNC_TO_LDS_B128 (ASYNCcnt), bias + leaky-relu
// ---------------------------------------------------------------------------

typedef __attribute__((ext_vector_type(16))) _Float16     v16h;
typedef __attribute__((ext_vector_type(8)))  float        v8f;
typedef __attribute__((ext_vector_type(8)))  unsigned int v8u;
typedef __attribute__((ext_vector_type(4)))  unsigned int v4u;
typedef __attribute__((ext_vector_type(4)))  int          v4i;

typedef __attribute__((address_space(1))) v4i glb_v4i;   // global src
typedef __attribute__((address_space(3))) v4i lds_v4i;   // LDS dst

#define LRELU(v) ((v) > 0.0f ? (v) : 0.01f * (v))

constexpr int BSZ  = 65536;
constexpr int KF   = 448;    // 425 padded to 32-multiple
constexpr int KH   = 288;    // 272 padded to 32-multiple
constexpr int KC   = 960;    // 64*15, exact
constexpr int NOUT = 1600;

// LDS row strides (f16 elements), padded to break bank conflicts
constexpr int SF = 456;      // 448 + 8
constexpr int SH = 296;      // 288 + 8
constexpr int SA3 = 40;      // 32 + 8  (80B rows: 16B-aligned for b128 traffic)
constexpr int SB3 = 40;

// ---- CDNA5 async memory->LDS copy (tracked with ASYNCcnt) ------------------
__device__ __forceinline__ void cp_async16(_Float16* lds, const _Float16* gl) {
  __builtin_amdgcn_global_load_async_to_lds_b128(
      (glb_v4i*)gl, (lds_v4i*)lds, 0 /*imm offset*/, 0 /*cpol*/);
}

// ---- WMMA fragment loaders -------------------------------------------------
// A (16x32 f16, M x K): lane m=lane&15, half=lane>>4;
// pairs p=0..7 map to K0 = 2*p + 16*half; dword idx = p
// -> one contiguous 32B run per lane (matches B: K = 16*half + e).
__device__ __forceinline__ v16h frag_a_lds(const _Float16* base, int stride,
                                           int lane, int kb) {
  const int m = lane & 15, lh = lane >> 4;
  const unsigned int* rp = (const unsigned int*)(base + m * stride + kb + 16 * lh);
  v8u t;
#pragma unroll
  for (int p = 0; p < 8; ++p)
    t[p] = rp[p];
  return __builtin_bit_cast(v16h, t);
}

// B (32x16 f16, K x N): lane n=lane&15 holds column n; K = 16*half + e contiguous.
// Global weights stored [N][K] row-major -> 32B contiguous per lane.
__device__ __forceinline__ v16h frag_b_gl(const _Float16* W, int ldk, int col,
                                          int lane, int kb) {
  const int n = lane & 15, lh = lane >> 4;
  return *(const v16h*)(W + (size_t)(col + n) * ldk + kb + 16 * lh);
}

// Same layout from LDS ([n][k] rows, stride SB3): two 16B loads (rows are 80B).
__device__ __forceinline__ v16h frag_b_lds(const _Float16* base, int stride,
                                           int lane) {
  const int n = lane & 15, lh = lane >> 4;
  const _Float16* p = base + n * stride + 16 * lh;
  v4u x0 = *(const v4u*)p;
  v4u x1 = *(const v4u*)(p + 8);
  v8u t;
  t[0] = x0[0]; t[1] = x0[1]; t[2] = x0[2]; t[3] = x0[3];
  t[4] = x1[0]; t[5] = x1[1]; t[6] = x1[2]; t[7] = x1[3];
  return __builtin_bit_cast(v16h, t);
}

#define WMMA_F16(a, b, c)                                                     \
  __builtin_amdgcn_wmma_f32_16x16x32_f16(false, (a), false, (b), (short)0,    \
                                         (c), false, false)

// ---------------------------------------------------------------------------
// k0: weight conversion / padding / re-layout to f16
// ---------------------------------------------------------------------------
__global__ __launch_bounds__(256) void k0_prep(
    const float* __restrict__ W_pf, const float* __restrict__ W_ph,
    const float* __restrict__ W_pv, const float* __restrict__ W_out,
    _Float16* __restrict__ Wpf16, _Float16* __restrict__ Wph16,
    _Float16* __restrict__ Wpv16, _Float16* __restrict__ Wout16) {
  int idx = blockIdx.x * 256 + threadIdx.x;
  const int NW0 = NOUT * KC;   // W_out  : 1,536,000
  const int NW1 = 320 * KF;    // W_pf   :   143,360
  const int NW2 = 64 * KH;     // W_ph/pv:    18,432
  if (idx < NW0) {
    const int o = idx / KC, kk = idx % KC;
    const int l = kk >> 6, c = kk & 63;          // kk = l*64 + c
    Wout16[idx] = (_Float16)W_out[(size_t)o * 960 + c * 15 + l];
  } else if ((idx -= NW0) < NW1) {
    const int q = idx / KF, k = idx % KF;
    Wpf16[idx] = (_Float16)(k < 425 ? W_pf[q * 425 + k] : 0.0f);
  } else if ((idx -= NW1) < NW2) {
    const int o = idx / KH, k = idx % KH;
    Wph16[idx] = (_Float16)(k < 272 ? W_ph[o * 272 + k] : 0.0f);
  } else if ((idx -= NW2) < NW2) {
    const int o = idx / KH, k = idx % KH;
    Wpv16[idx] = (_Float16)(k < 272 ? W_pv[o * 272 + k] : 0.0f);
  }
}

// ---------------------------------------------------------------------------
// k1: 16 samples per WG.  Gather (one-hot depthwise) into LDS f16, bias+lrelu,
//     then WMMA pointwise GEMMs; hori/vert share one LDS buffer sequentially.
// ---------------------------------------------------------------------------
__global__ __launch_bounds__(256) void k1_gather_stage1(
    const int* __restrict__ x,
    const float* __restrict__ W_df, const float* __restrict__ b_df,
    const float* __restrict__ b_pf,
    const float* __restrict__ W_dh, const float* __restrict__ b_dh,
    const float* __restrict__ b_ph,
    const float* __restrict__ W_dv, const float* __restrict__ b_dv,
    const float* __restrict__ b_pv,
    const _Float16* __restrict__ Wpf16, const _Float16* __restrict__ Wph16,
    const _Float16* __restrict__ Wpv16, _Float16* __restrict__ cat) {
  __shared__ _Float16 Af[16 * SF];     // 14.25 KB
  __shared__ _Float16 Ahv[80 * SH];    // 46.25 KB  (hori, then reused for vert)

  const int tid = threadIdx.x, lane = tid & 31, wave = tid >> 5;
  const int bBase = blockIdx.x * 16;
  const int n = lane & 15, lh = lane >> 4;

  // ---- zero both LDS matrices (padding must be 0) ----
  for (int i = tid; i < 16 * SF; i += 256) Af[i] = (_Float16)0.0f;
  for (int i = tid; i < 80 * SH; i += 256) Ahv[i] = (_Float16)0.0f;
  __syncthreads();

  // ---- gather: full + hori (each wave owns 2 samples) ----
  for (int ss = 0; ss < 2; ++ss) {
    const int s = wave * 2 + ss;
    const int* xb = x + (size_t)(bBase + s) * 25;
    _Float16* fr = Af + s * SF;
    _Float16* hr = Ahv + s * 5 * SH;
#pragma unroll 1
    for (int p = 0; p < 25; ++p) {
      const int c = xb[p];
      const int r = p / 5, j = p % 5;
      if (lane < 25) {   // full: k = c*25 + m, add W_df[c][m][p]
        const int idx = c * 25 + lane;
        fr[idx] = fr[idx] + (_Float16)W_df[idx * 25 + p];
      }
      if (lane < 16) {   // hori: k = c*16 + m at row r, add W_dh[c][m][j]
        const int kk = c * 16 + lane;
        hr[r * SH + kk] = hr[r * SH + kk] + (_Float16)W_dh[c * 80 + lane * 5 + j];
      }
    }
    for (int k = lane; k < 425; k += 32) {
      float v = (float)fr[k] + b_df[k];
      fr[k] = (_Float16)LRELU(v);
    }
    for (int r = 0; r < 5; ++r)
      for (int k = lane; k < 272; k += 32) {
        float v = (float)hr[r * SH + k] + b_dh[k];
        hr[r * SH + k] = (_Float16)LRELU(v);
      }
  }
  __syncthreads();

  // ---- GEMM full: [16 x 448] x [448 x 320] -> cat[:, u*64 + c] ----
  for (int t = wave; t < 20; t += 8) {
    v8f acc = {};
#pragma unroll 2
    for (int ks = 0; ks < KF / 32; ++ks) {
      v16h a = frag_a_lds(Af, SF, lane, ks * 32);
      v16h b = frag_b_gl(Wpf16, KF, t * 16, lane, ks * 32);
      acc = WMMA_F16(a, b, acc);
    }
    const int q = t * 16 + n;
    const int c = q / 5, u = q % 5;
    const float bias = b_pf[q];
#pragma unroll
    for (int i = 0; i < 8; ++i) {
      const int s = i + 8 * lh;
      float v = acc[i] + bias;
      cat[(size_t)(bBase + s) * KC + u * 64 + c] = (_Float16)LRELU(v);
    }
  }

  // ---- GEMM hori: [80 x 288] x [288 x 64] -> cat[:, (5+r)*64 + o] ----
  for (int pt = wave; pt < 20; pt += 8) {
    const int mt = pt >> 2, nt = pt & 3;
    v8f acc = {};
#pragma unroll 3
    for (int ks = 0; ks < KH / 32; ++ks) {
      v16h a = frag_a_lds(Ahv + mt * 16 * SH, SH, lane, ks * 32);
      v16h b = frag_b_gl(Wph16, KH, nt * 16, lane, ks * 32);
      acc = WMMA_F16(a, b, acc);
    }
    const int o = nt * 16 + n;
    const float bias = b_ph[o];
#pragma unroll
    for (int i = 0; i < 8; ++i) {
      const int row = mt * 16 + i + 8 * lh;   // = s*5 + r
      const int s = row / 5, r = row % 5;
      float v = acc[i] + bias;
      cat[(size_t)(bBase + s) * KC + (5 + r) * 64 + o] = (_Float16)LRELU(v);
    }
  }
  __syncthreads();   // everyone done reading Ahv (hori) before reuse

  // ---- vert: re-zero, re-gather, GEMM ----
  for (int i = tid; i < 80 * SH; i += 256) Ahv[i] = (_Float16)0.0f;
  __syncthreads();

  for (int ss = 0; ss < 2; ++ss) {
    const int s = wave * 2 + ss;
    const int* xb = x + (size_t)(bBase + s) * 25;
    _Float16* vr = Ahv + s * 5 * SH;
#pragma unroll 1
    for (int p = 0; p < 25; ++p) {
      const int c = xb[p];
      const int r = p / 5, j = p % 5;
      if (lane < 16) {   // vert: k = c*16 + m at column j, add W_dv[c][m][r]
        const int kk = c * 16 + lane;
        vr[j * SH + kk] = vr[j * SH + kk] + (_Float16)W_dv[c * 80 + lane * 5 + r];
      }
    }
    for (int j = 0; j < 5; ++j)
      for (int k = lane; k < 272; k += 32) {
        float v = (float)vr[j * SH + k] + b_dv[k];
        vr[j * SH + k] = (_Float16)LRELU(v);
      }
  }
  __syncthreads();

  for (int pt = wave; pt < 20; pt += 8) {
    const int mt = pt >> 2, nt = pt & 3;
    v8f acc = {};
#pragma unroll 3
    for (int ks = 0; ks < KH / 32; ++ks) {
      v16h a = frag_a_lds(Ahv + mt * 16 * SH, SH, lane, ks * 32);
      v16h b = frag_b_gl(Wpv16, KH, nt * 16, lane, ks * 32);
      acc = WMMA_F16(a, b, acc);
    }
    const int o = nt * 16 + n;
    const float bias = b_pv[o];
#pragma unroll
    for (int i = 0; i < 8; ++i) {
      const int row = mt * 16 + i + 8 * lh;   // = s*5 + j
      const int s = row / 5, j = row % 5;
      float v = acc[i] + bias;
      cat[(size_t)(bBase + s) * KC + (10 + j) * 64 + o] = (_Float16)LRELU(v);
    }
  }
}

// ---------------------------------------------------------------------------
// k3: out = lrelu(cat[B,960] x Wout16[960,1600]^T-layout + b_out), f32.
//     WG tile 128(M) x 64(N); 8 waves = 4(M) x 2(N), 2x2 frags each.
//     Double-buffered staging via GLOBAL_LOAD_ASYNC_TO_LDS_B128:
//       wait ASYNCcnt==0 -> barrier -> issue next buffer -> compute current.
//     One barrier per K-step is sufficient (issuers of buf 1-p only run after
//     the barrier that followed all readers' compute of buf 1-p).
// ---------------------------------------------------------------------------
__global__ __launch_bounds__(256) void k3_gemm_out(
    const _Float16* __restrict__ cat, const _Float16* __restrict__ Wout16,
    const float* __restrict__ b_out, float* __restrict__ out) {
  __shared__ _Float16 As[2][128 * SA3];   // 2 x 10 KB
  __shared__ _Float16 Bs[2][64 * SB3];    // 2 x  5 KB

  const int tid = threadIdx.x, lane = tid & 31, wave = tid >> 5;
  const int wm = wave >> 1, wn = wave & 1;
  const int mBase = blockIdx.y * 128;
  const int nBase = blockIdx.x * 64;

  // staging: A = 128 rows x 64B (each thread: 2 x 16B async);
  //          B =  64 rows x 64B (each thread: 1 x 16B async)
  const int arow = tid >> 1, ah = tid & 1;
  const int brow = tid >> 2, bq = tid & 3;

  const _Float16* gA = cat + (size_t)(mBase + arow) * KC + ah * 16;
  const _Float16* gB = Wout16 + (size_t)(nBase + brow) * KC + bq * 8;

  // prologue: async-stage K-step 0 into buffer 0
  {
    _Float16* la = &As[0][arow * SA3 + ah * 16];
    cp_async16(la, gA);
    cp_async16(la + 8, gA + 8);
    cp_async16(&Bs[0][brow * SB3 + bq * 8], gB);
  }

  v8f acc00 = {}, acc01 = {}, acc10 = {}, acc11 = {};

  for (int ks = 0; ks < KC / 32; ++ks) {
    const int p = ks & 1;
    __builtin_amdgcn_s_wait_asynccnt(0);   // my async copies have landed in LDS
    __syncthreads();                       // everyone's copies have landed
    if (ks + 1 < KC / 32) {                // issue next K-step into other buffer
      const int kb = (ks + 1) * 32;
      _Float16* la = &As[p ^ 1][arow * SA3 + ah * 16];
      cp_async16(la, gA + kb);
      cp_async16(la + 8, gA + kb + 8);
      cp_async16(&Bs[p ^ 1][brow * SB3 + bq * 8], gB + kb);
    }
    v16h a0 = frag_a_lds(&As[p][(wm * 32 + 0) * SA3], SA3, lane, 0);
    v16h a1 = frag_a_lds(&As[p][(wm * 32 + 16) * SA3], SA3, lane, 0);
    v16h b0 = frag_b_lds(&Bs[p][(wn * 32 + 0) * SB3], SB3, lane);
    v16h b1 = frag_b_lds(&Bs[p][(wn * 32 + 16) * SB3], SB3, lane);
    acc00 = WMMA_F16(a0, b0, acc00);
    acc01 = WMMA_F16(a0, b1, acc01);
    acc10 = WMMA_F16(a1, b0, acc10);
    acc11 = WMMA_F16(a1, b1, acc11);
  }

  // epilogue: bias + leaky-relu, f32 store
  const int n = lane & 15, lh = lane >> 4;
  const v8f* accs[4] = {&acc00, &acc01, &acc10, &acc11};
#pragma unroll
  for (int f = 0; f < 4; ++f) {
    const int fm = f >> 1, fn = f & 1;
    const int col = nBase + wn * 32 + fn * 16 + n;
    const float bias = b_out[col];
    const v8f a = *accs[f];
#pragma unroll
    for (int i = 0; i < 8; ++i) {
      const int row = mBase + wm * 32 + fm * 16 + i + 8 * lh;
      float v = a[i] + bias;
      out[(size_t)row * NOUT + col] = LRELU(v);
    }
  }
}

// ---------------------------------------------------------------------------
extern "C" void kernel_launch(void* const* d_in, const int* in_sizes, int n_in,
                              void* d_out, int out_size, void* d_ws,
                              size_t ws_size, hipStream_t stream) {
  const int*   x     = (const int*)d_in[0];
  const float* W_df  = (const float*)d_in[1];
  const float* b_df  = (const float*)d_in[2];
  const float* W_pf  = (const float*)d_in[3];
  const float* b_pf  = (const float*)d_in[4];
  const float* W_dh  = (const float*)d_in[5];
  const float* b_dh  = (const float*)d_in[6];
  const float* W_ph  = (const float*)d_in[7];
  const float* b_ph  = (const float*)d_in[8];
  const float* W_dv  = (const float*)d_in[9];
  const float* b_dv  = (const float*)d_in[10];
  const float* W_pv  = (const float*)d_in[11];
  const float* b_pv  = (const float*)d_in[12];
  const float* W_out = (const float*)d_in[13];
  const float* b_out = (const float*)d_in[14];
  float* out = (float*)d_out;

  char* ws = (char*)d_ws;
  auto alignup = [](size_t v) { return (v + 255) & ~(size_t)255; };
  size_t off = 0;
  _Float16* cat    = (_Float16*)(ws + off); off = alignup(off + (size_t)BSZ * KC * 2);
  _Float16* Wout16 = (_Float16*)(ws + off); off = alignup(off + (size_t)NOUT * KC * 2);
  _Float16* Wpf16  = (_Float16*)(ws + off); off = alignup(off + (size_t)320 * KF * 2);
  _Float16* Wph16  = (_Float16*)(ws + off); off = alignup(off + (size_t)64 * KH * 2);
  _Float16* Wpv16  = (_Float16*)(ws + off); off = alignup(off + (size_t)64 * KH * 2);

  // k0: weight prep
  {
    const int total = NOUT * KC + 320 * KF + 2 * 64 * KH;
    k0_prep<<<(total + 255) / 256, 256, 0, stream>>>(
        W_pf, W_ph, W_pv, W_out, Wpf16, Wph16, Wpv16, Wout16);
  }
  // k1: gather + stage-1 GEMMs -> cat (f16, l-major K layout)
  k1_gather_stage1<<<BSZ / 16, 256, 0, stream>>>(
      x, W_df, b_df, b_pf, W_dh, b_dh, b_ph, W_dv, b_dv, b_pv,
      Wpf16, Wph16, Wpv16, cat);
  // k3: final GEMM + bias + lrelu
  {
    dim3 grid(NOUT / 64, BSZ / 128);
    k3_gemm_out<<<grid, 256, 0, stream>>>(cat, Wout16, b_out, out);
  }
}
